// Wide_48034914238966
// MI455X (gfx1250) — matrix operand order repaired and checked by reading, not compiled, not run on hardware
//
#include <hip/hip_runtime.h>
#include <hip/hip_fp16.h>
#include <math.h>

typedef __attribute__((ext_vector_type(16))) _Float16 v16h;
typedef __attribute__((ext_vector_type(8)))  _Float16 v8h;
typedef __attribute__((ext_vector_type(4)))  _Float16 v4h;
typedef __attribute__((ext_vector_type(8)))  float    v8f;

#define BATCH_N   131072
#define EMB_N     128
#define ROW_H     168              // halves per padded row (336 B stride: conflict-free for 16-row A fetch)
#define NWAVES    4                // waves per workgroup
#define ROWS_W    16               // batch rows per wave (one WMMA M-tile)
#define LDS_HALVES (NWAVES * 3 * ROWS_W * ROW_H)   // 64512 B

__device__ __forceinline__ float fast_tanhf(float x) {
#if defined(__AMDGCN__) && __has_builtin(__builtin_amdgcn_tanhf)
  return __builtin_amdgcn_tanhf(x);        // gfx1250 v_tanh_f32
#else
  return tanhf(x);
#endif
}

__device__ __forceinline__ void load_frags(const _Float16* rowU, const _Float16* rowP,
                                           const _Float16* rowN, int wbase, int koff,
                                           v16h& au, v16h& ap, v16h& an) {
  const v8h u0 = *(const v8h*)&rowU[wbase + koff];
  const v8h u1 = *(const v8h*)&rowU[wbase + koff + 16];
  const v8h p0 = *(const v8h*)&rowP[wbase + koff];
  const v8h p1 = *(const v8h*)&rowP[wbase + koff + 16];
  const v8h n0 = *(const v8h*)&rowN[wbase + koff];
  const v8h n1 = *(const v8h*)&rowN[wbase + koff + 16];
  #pragma unroll
  for (int e = 0; e < 8; ++e) {
    au[e] = u0[e]; au[e + 8] = u1[e];
    ap[e] = p0[e]; ap[e + 8] = p1[e];
    an[e] = n0[e]; an[e + 8] = n1[e];
  }
}

__global__ __launch_bounds__(NWAVES * 32)
void wide_ranker_kernel(const int* __restrict__ users,
                        const int* __restrict__ positives,
                        const int* __restrict__ negatives,
                        const float* __restrict__ user_emb,
                        const float* __restrict__ item_emb,
                        const float* __restrict__ w1, const float* __restrict__ b1,
                        const float* __restrict__ w2, const float* __restrict__ b2,
                        const float* __restrict__ w3, const float* __restrict__ b3,
                        const float* __restrict__ w4, const float* __restrict__ b4,
                        const float* __restrict__ w5, const float* __restrict__ b5,
                        const float* __restrict__ lin_w, const float* __restrict__ lin_b,
                        float* __restrict__ out)
{
  __shared__ _Float16 lds[LDS_HALVES];

  const int tid   = threadIdx.x;
  const int wave  = tid >> 5;
  const int lane  = tid & 31;
  const int laneN = lane & 15;     // N column (B/C layout) and M row (A layout)
  const int hi    = lane >> 4;     // half-wave select
  const int row_base = (blockIdx.x * NWAVES + wave) * ROWS_W;

  _Float16* my = &lds[wave * (3 * ROWS_W * ROW_H)];

  const float* ws[5] = {w1, w2, w3, w4, w5};
  const int    ks[5] = {1, 2, 4, 8, 16};
  float bs[5] = {b1[0], b2[0], b3[0], b4[0], b5[0]};
  float lw[5];
  #pragma unroll
  for (int i = 0; i < 5; ++i) lw[i] = lin_w[i];
  const float lb = lin_b[0];

  // Banded 32x16 f16 B-fragments, one per conv branch; constant across tiles.
  // 16-bit B layout: lane = N (mod 16), element e of v16h = K = hi*8 + e (+8 if e >= 8).
  // B[K][N] = w_k[K - N] when 0 <= K-N < k, else 0.
  v16h bfrag[5];
  #pragma unroll
  for (int i = 0; i < 5; ++i) {
    v16h bf;
    #pragma unroll
    for (int e = 0; e < 16; ++e) {
      const int r = hi * 8 + e + (e >= 8 ? 8 : 0) - laneN;
      const float v = (r >= 0 && r < ks[i]) ? ws[i][r] : 0.0f;
      bf[e] = (_Float16)v;
    }
    bfrag[i] = bf;
  }

  // Gather embedding rows, convert fp32 -> fp16, stage into LDS with zero pad
  // so tile 7's K-window [112,144) never reads garbage.
  const int u_l = users[row_base + laneN];       // lane l<16 owns row l's indices
  const int p_l = positives[row_base + laneN];
  const int n_l = negatives[row_base + laneN];
  for (int m = 0; m < ROWS_W; ++m) {
    const int ui = __shfl(u_l, m, 32);
    const int pi = __shfl(p_l, m, 32);
    const int ni = __shfl(n_l, m, 32);
    const float4 fu = ((const float4*)(user_emb + (size_t)ui * EMB_N))[lane];
    const float4 fp = ((const float4*)(item_emb + (size_t)pi * EMB_N))[lane];
    const float4 fn = ((const float4*)(item_emb + (size_t)ni * EMB_N))[lane];
    const v4h hu = {(_Float16)fu.x, (_Float16)fu.y, (_Float16)fu.z, (_Float16)fu.w};
    const v4h hp = {(_Float16)fp.x, (_Float16)fp.y, (_Float16)fp.z, (_Float16)fp.w};
    const v4h hn = {(_Float16)fn.x, (_Float16)fn.y, (_Float16)fn.z, (_Float16)fn.w};
    *(v4h*)&my[(0 * ROWS_W + m) * ROW_H + lane * 4] = hu;
    *(v4h*)&my[(1 * ROWS_W + m) * ROW_H + lane * 4] = hp;
    *(v4h*)&my[(2 * ROWS_W + m) * ROW_H + lane * 4] = hn;
    if (lane < 10) {                   // zero halves 128..167 (40 halves)
      const v4h z = {};
      *(v4h*)&my[(0 * ROWS_W + m) * ROW_H + EMB_N + lane * 4] = z;
      *(v4h*)&my[(1 * ROWS_W + m) * ROW_H + EMB_N + lane * 4] = z;
      *(v4h*)&my[(2 * ROWS_W + m) * ROW_H + EMB_N + lane * 4] = z;
    }
  }
  // Per-wave LDS RAW fence (no cross-wave sharing, so no workgroup barrier).
  asm volatile("s_wait_dscnt 0" ::: "memory");

  v8f accp = {};   // sum_k lw[k] * tanh(cu)*tanh(cp), per (M,N) element
  v8f accn = {};

  // 16-bit A layout: lane holds row M = lane&15; halves [koff,koff+8) and
  // [koff+16,koff+24) of the K-window, koff = 8*hi.
  const _Float16* rowU = &my[(0 * ROWS_W + laneN) * ROW_H];
  const _Float16* rowP = &my[(1 * ROWS_W + laneN) * ROW_H];
  const _Float16* rowN = &my[(2 * ROWS_W + laneN) * ROW_H];
  const int koff = hi * 8;
  const v8f cz = {};

  // ---- software pipeline: issue WMMAs for branch i+1 / tile t+1 before ----
  // ---- consuming branch i, so the F16-WMMA->VALU hazard (4 co-exec) is ----
  // ---- always covered by ~80 independent VALU/trans ops.               ----
  v16h auA, apA, anA, auB, apB, anB;
  load_frags(rowU, rowP, rowN, 0, koff, auA, apA, anA);

  v8f cu = __builtin_amdgcn_wmma_f32_16x16x32_f16(false, auA, false, bfrag[0], (short)0, cz, false, false);
  v8f cp = __builtin_amdgcn_wmma_f32_16x16x32_f16(false, apA, false, bfrag[0], (short)0, cz, false, false);
  v8f cn = __builtin_amdgcn_wmma_f32_16x16x32_f16(false, anA, false, bfrag[0], (short)0, cz, false, false);
  float pcoef = (laneN <= (EMB_N - ks[0])) ? lw[0] : 0.0f;   // tile 0, branch 0
  float pbias = bs[0];

  for (int t = 0; t < 8; ++t) {
    const bool last_t = (t == 7);
    #pragma unroll
    for (int i = 0; i < 5; ++i) {
      if (i == 0 && !last_t)   // prefetch next tile's A-fragments early
        load_frags(rowU, rowP, rowN, (t + 1) * 16, koff, auB, apB, anB);

      v8f ncu, ncp, ncn;
      float ncoef = 0.0f, nbias = 0.0f;
      const bool has_next = !(last_t && i == 4);
      if (i < 4) {
        ncu = __builtin_amdgcn_wmma_f32_16x16x32_f16(false, auA, false, bfrag[i + 1], (short)0, cz, false, false);
        ncp = __builtin_amdgcn_wmma_f32_16x16x32_f16(false, apA, false, bfrag[i + 1], (short)0, cz, false, false);
        ncn = __builtin_amdgcn_wmma_f32_16x16x32_f16(false, anA, false, bfrag[i + 1], (short)0, cz, false, false);
        ncoef = ((t * 16 + laneN) <= (EMB_N - ks[i + 1])) ? lw[i + 1] : 0.0f;
        nbias = bs[i + 1];
      } else if (!last_t) {
        ncu = __builtin_amdgcn_wmma_f32_16x16x32_f16(false, auB, false, bfrag[0], (short)0, cz, false, false);
        ncp = __builtin_amdgcn_wmma_f32_16x16x32_f16(false, apB, false, bfrag[0], (short)0, cz, false, false);
        ncn = __builtin_amdgcn_wmma_f32_16x16x32_f16(false, anB, false, bfrag[0], (short)0, cz, false, false);
        ncoef = (((t + 1) * 16 + laneN) <= (EMB_N - ks[0])) ? lw[0] : 0.0f;
        nbias = bs[0];
      }

      // consume pending branch results
      #pragma unroll
      for (int e = 0; e < 8; ++e) {
        const float tu = fast_tanhf(cu[e] + pbias);
        const float tp = fast_tanhf(cp[e] + pbias);
        const float tn = fast_tanhf(cn[e] + pbias);
        accp[e] += pcoef * (tu * tp);
        accn[e] += pcoef * (tu * tn);
      }

      if (has_next) { cu = ncu; cp = ncp; cn = ncn; pcoef = ncoef; pbias = nbias; }
      if (i == 4)   { auA = auB; apA = apB; anA = anB; }
    }
  }

  // Reduce over the 16 N-columns inside each half-wave (xor < 16 keeps halves separate).
  #pragma unroll
  for (int m = 1; m < 16; m <<= 1) {
    #pragma unroll
    for (int e = 0; e < 8; ++e) {
      accp[e] += __shfl_xor(accp[e], m, 32);
      accn[e] += __shfl_xor(accn[e], m, 32);
    }
  }

  // C layout: VGPR e holds M=e (lanes 0-15) / M=8+e (lanes 16-31).
  if (laneN == 0) {
    const int rb = row_base + hi * 8;
    #pragma unroll
    for (int e = 0; e < 8; ++e) {
      out[rb + e]           = accp[e] + lb;   // p_score
      out[BATCH_N + rb + e] = accn[e] + lb;   // n_score
    }
  }
}

extern "C" void kernel_launch(void* const* d_in, const int* in_sizes, int n_in,
                              void* d_out, int out_size, void* d_ws, size_t ws_size,
                              hipStream_t stream) {
  (void)in_sizes; (void)n_in; (void)out_size; (void)d_ws; (void)ws_size;
  const int*   users     = (const int*)d_in[0];
  const int*   positives = (const int*)d_in[1];
  const int*   negatives = (const int*)d_in[2];
  const float* user_emb  = (const float*)d_in[3];
  const float* item_emb  = (const float*)d_in[4];
  const float* w1 = (const float*)d_in[5];  const float* b1 = (const float*)d_in[6];
  const float* w2 = (const float*)d_in[7];  const float* b2 = (const float*)d_in[8];
  const float* w3 = (const float*)d_in[9];  const float* b3 = (const float*)d_in[10];
  const float* w4 = (const float*)d_in[11]; const float* b4 = (const float*)d_in[12];
  const float* w5 = (const float*)d_in[13]; const float* b5 = (const float*)d_in[14];
  const float* lin_w = (const float*)d_in[15];
  const float* lin_b = (const float*)d_in[16];
  float* out = (float*)d_out;

  const int rows_per_block = NWAVES * ROWS_W;          // 64
  dim3 grid(BATCH_N / rows_per_block);                 // 2048
  dim3 block(NWAVES * 32);                             // 128
  hipLaunchKernelGGL(wide_ranker_kernel, grid, block, 0, stream,
                     users, positives, negatives, user_emb, item_emb,
                     w1, b1, w2, b2, w3, b3, w4, b4, w5, b5,
                     lin_w, lin_b, out);
}
